// Wavernn_para_86990267613892
// MI455X (gfx1250) — matrix operand order repaired and checked by reading, not compiled, not run on hardware
//
#include <hip/hip_runtime.h>
#include <hip/hip_bf16.h>
#include <stdint.h>

#define BB 64
#define LL 2048
#define CC 20
#define H1 384
#define G1 1152   /* 3*H1 */
#define H2 16
#define G2 48
#define FF 20
#define G3 60
#define NB1 24    /* persistent blocks for GRU1: 24 * 16 cols = 384 */

typedef __attribute__((ext_vector_type(16))) _Float16     v16h;
typedef __attribute__((ext_vector_type(8)))  float        v8f;
typedef __attribute__((ext_vector_type(8)))  unsigned int v8u;

// ---------------------------------------------------------------------------
// Tile loaders matching the CDNA5 16-bit WMMA VGPR layouts (CDNA5 ISA)
// A (16xK=32, M rows): lanes 0-15 -> M=lane, K{0..7,16..23}; lanes 16-31 -> K{8..15,24..31}
// ---------------------------------------------------------------------------
__device__ __forceinline__ v16h load_tileA(const _Float16* __restrict base,
                                           int row0, int ldk, int k0, int lane) {
  const int r  = lane & 15;
  const int hi = lane >> 4;
  const unsigned int* q =
      (const unsigned int*)(base + (size_t)(row0 + r) * ldk + k0 + hi * 8);
  v8u t;
  t[0] = q[0]; t[1] = q[1]; t[2] = q[2]; t[3] = q[3];   // K 0..7  (+8*hi)
  t[4] = q[8]; t[5] = q[9]; t[6] = q[10]; t[7] = q[11]; // K 16..23 (+8*hi)
  return __builtin_bit_cast(v16h, t);
}

// B (K=32 x 16, row-major source indexed [n][k]):
// lanes 0-15 -> N=lane, K{0..7,16..23}; lanes 16-31 -> N=lane-16, K{8..15,24..31}
// Works for both global and LDS-resident sources (address space inferred).
__device__ __forceinline__ v16h load_tileB(const _Float16* base,
                                           int n0, int ldk, int k0, int lane) {
  const int n  = lane & 15;
  const int hi = lane >> 4;
  const unsigned int* q =
      (const unsigned int*)(base + (size_t)(n0 + n) * ldk + k0 + hi * 8);
  v8u t;
  t[0] = q[0]; t[1] = q[1]; t[2] = q[2]; t[3] = q[3];
  t[4] = q[8]; t[5] = q[9]; t[6] = q[10]; t[7] = q[11];
  return __builtin_bit_cast(v16h, t);
}

__device__ __forceinline__ v8f wmma_f16(v16h a, v16h b, v8f c) {
  return __builtin_amdgcn_wmma_f32_16x16x32_f16(false, a, false, b, (short)0, c,
                                                false, false);
}

__device__ __forceinline__ float sigmoidf_(float x) {
  return 1.0f / (1.0f + __expf(-x));
}

// ---------------------------------------------------------------------------
// Conversions / init
// ---------------------------------------------------------------------------
__global__ void k_convert_x(const float* __restrict x, _Float16* __restrict x16) {
  // x: (B, L, C) fp32 -> x16: (L, B, 32) f16, zero-padded K
  size_t i = blockIdx.x * (size_t)blockDim.x + threadIdx.x;
  const size_t total = (size_t)LL * BB * 32;
  if (i >= total) return;
  int c = (int)(i & 31);
  size_t tb = i >> 5;
  int b = (int)(tb % BB);
  int t = (int)(tb / BB);
  float v = (c < CC) ? x[((size_t)b * LL + t) * CC + c] : 0.0f;
  x16[i] = (_Float16)v;
}

__global__ void k_convert_w(const float* __restrict w, _Float16* __restrict w16,
                            int rows, int kin, int kout) {
  size_t i = blockIdx.x * (size_t)blockDim.x + threadIdx.x;
  size_t total = (size_t)rows * kout;
  if (i >= total) return;
  int k = (int)(i % kout);
  int r = (int)(i / kout);
  float v = (k < kin) ? w[(size_t)r * kin + k] : 0.0f;
  w16[i] = (_Float16)v;
}

__global__ void k_init(const float* __restrict h1, float* __restrict h32,
                       _Float16* __restrict h16, unsigned* __restrict bar) {
  int i = blockIdx.x * blockDim.x + threadIdx.x;
  if (i < BB * H1) {
    float v = h1[i];
    h32[i] = v;          // buffer 0
    h16[i] = (_Float16)v;
  }
  if (i < 8) bar[i] = 0u;
}

// ---------------------------------------------------------------------------
// GRU1 persistent recurrence: 24 blocks x 128 threads (4 waves).
// Block b owns hidden columns [b*16, b*16+16) across all three gates.
// The 3-gate Whh slice (48 rows x 384 K, f16 = 36 KB) is staged in LDS once;
// per-step B operands come from ds_load_b128 instead of 2048x L2 re-reads.
// Wave w = M-tile (batch rows w*16 .. w*16+15).
// ---------------------------------------------------------------------------
__global__ void __launch_bounds__(128) k_gru1(
    const _Float16* __restrict x16,   // (L, B, 32)
    const _Float16* __restrict wih,   // (G1, 32)  f16
    const _Float16* __restrict whh,   // (G1, H1)  f16
    const float* __restrict bih, const float* __restrict bhh,
    float* __restrict h32,            // [2][B*H1]
    _Float16* __restrict h16,         // [2][B*H1]
    _Float16* __restrict y1,          // (L, B, H1) f16
    float* __restrict h1f,            // (B, H1) final state -> d_out
    unsigned* __restrict bar)         // bar[0]=arrive counter, bar[1]=generation
{
  __shared__ _Float16 sB[3 * 16 * H1];   // 36 KB: r|z|n gate rows for our slice

  const int lane = threadIdx.x & 31;
  const int wave = threadIdx.x >> 5;     // 0..3 -> M tile
  const int m0 = wave * 16;
  const int j0 = blockIdx.x * 16;
  const int ncol = j0 + (lane & 15);
  const int hi = lane >> 4;

  // stage loop-invariant Whh slice into LDS (16B chunks, coalesced)
  for (int idx = threadIdx.x; idx < 3 * 16 * (H1 / 8); idx += 128) {
    const int chunk = idx % (H1 / 8);   // 8-half (16B) chunk within row
    const int rowl  = idx / (H1 / 8);   // 0..47 = g*16 + n
    const int g = rowl >> 4;
    const int n = rowl & 15;
    const uint4* src =
        (const uint4*)(whh + (size_t)(g * H1 + j0 + n) * H1) + chunk;
    ((uint4*)(sB + (size_t)rowl * H1))[chunk] = *src;
  }
  __syncthreads();

  // hoist per-column biases out of the time loop
  const float b_r  = bih[ncol] + bhh[ncol];
  const float b_z  = bih[H1 + ncol] + bhh[H1 + ncol];
  const float b_in = bih[2 * H1 + ncol];
  const float b_hn = bhh[2 * H1 + ncol];

  // input-projection B tiles are loop-invariant: keep in registers
  const v16h bxr = load_tileB(wih, j0, 32, 0, lane);
  const v16h bxz = load_tileB(wih, H1 + j0, 32, 0, lane);
  const v16h bxn = load_tileB(wih, 2 * H1 + j0, 32, 0, lane);

  for (int t = 0; t < LL; ++t) {
    const int cur = t & 1;
    const float*    hc32 = h32 + (size_t)cur * BB * H1;
    const _Float16* hc16 = h16 + (size_t)cur * BB * H1;
    float*    hn32 = h32 + (size_t)(cur ^ 1) * BB * H1;
    _Float16* hn16 = h16 + (size_t)(cur ^ 1) * BB * H1;

    v8f ar = {}, az = {}, anx = {}, anh = {};

    // input projection (K padded 20->32, one WMMA step per gate)
    {
      v16h a = load_tileA(x16 + (size_t)t * BB * 32, m0, 32, 0, lane);
      ar  = wmma_f16(a, bxr, ar);
      az  = wmma_f16(a, bxz, az);
      anx = wmma_f16(a, bxn, anx);
    }
    // hidden projection: K = 384 = 12 WMMA steps per gate; B from LDS
    for (int kk = 0; kk < H1; kk += 32) {
      v16h a  = load_tileA(hc16, m0, H1, kk, lane);
      v16h br = load_tileB(sB,               0, H1, kk, lane);
      v16h bz = load_tileB(sB + 16 * H1,     0, H1, kk, lane);
      v16h bn = load_tileB(sB + 2 * 16 * H1, 0, H1, kk, lane);
      ar  = wmma_f16(a, br, ar);
      az  = wmma_f16(a, bz, az);
      anh = wmma_f16(a, bn, anh);
    }

    // gates + state update (C/D layout: lane l, vgpr v -> M=m0+v+8*hi, N=ncol)
#pragma unroll
    for (int v = 0; v < 8; ++v) {
      const int m = m0 + v + 8 * hi;
      const float hold = hc32[(size_t)m * H1 + ncol];
      const float r = sigmoidf_(ar[v] + b_r);
      const float z = sigmoidf_(az[v] + b_z);
      const float n = tanhf(anx[v] + b_in + r * (anh[v] + b_hn));
      const float hnew = (1.0f - z) * n + z * hold;
      hn32[(size_t)m * H1 + ncol] = hnew;
      const _Float16 hh = (_Float16)hnew;
      hn16[(size_t)m * H1 + ncol] = hh;
      // y1 is written once, read once later: keep it out of the cache set
      __builtin_nontemporal_store(hh, &y1[((size_t)t * BB + m) * H1 + ncol]);
    }

    // grid-wide step barrier (24 resident blocks; generation counter)
    __syncthreads();
    if (threadIdx.x == 0) {
      __threadfence();
      unsigned old = atomicAdd(&bar[0], 1u);
      if (old == NB1 - 1) {
        atomicExch(&bar[0], 0u);
        __threadfence();
        atomicExch(&bar[1], (unsigned)(t + 1));
      } else {
        while (__hip_atomic_load(&bar[1], __ATOMIC_ACQUIRE,
                                 __HIP_MEMORY_SCOPE_AGENT) < (unsigned)(t + 1)) {
          __builtin_amdgcn_s_sleep(1);
        }
      }
    }
    __syncthreads();
  }

  // LL is even -> final state sits in buffer 0
  for (int idx = threadIdx.x; idx < BB * 16; idx += blockDim.x) {
    const int b = idx >> 4;
    const int j = j0 + (idx & 15);
    h1f[(size_t)b * H1 + j] = h32[(size_t)b * H1 + j];
  }
}

// ---------------------------------------------------------------------------
// Bulk WMMA GEMM: xp2 = y1 (M=L*B, K=384) x Wih2^T (N=48) + bih2
// Wih2 (36 KB) staged in LDS once per block; A streams from global.
// ---------------------------------------------------------------------------
__global__ void __launch_bounds__(256) k_xp2(
    const _Float16* __restrict y1,  // (M, 384)
    const _Float16* __restrict w16, // (48, 384)
    const float* __restrict bih,    // (48)
    float* __restrict xp2)          // (M, 48)
{
  __shared__ _Float16 sW[G2 * H1];  // 36 KB
  for (int idx = threadIdx.x; idx < G2 * H1 / 8; idx += 256) {
    ((uint4*)sW)[idx] = ((const uint4*)w16)[idx];
  }
  __syncthreads();

  const int lane = threadIdx.x & 31;
  const int wave = threadIdx.x >> 5;
  const int m0 = blockIdx.x * 128 + wave * 16;
  v8f acc0 = {}, acc1 = {}, acc2 = {};
  for (int kk = 0; kk < H1; kk += 32) {
    v16h a  = load_tileA(y1, m0, H1, kk, lane);
    v16h b0 = load_tileB(sW, 0, H1, kk, lane);
    v16h b1 = load_tileB(sW, 16, H1, kk, lane);
    v16h b2 = load_tileB(sW, 32, H1, kk, lane);
    acc0 = wmma_f16(a, b0, acc0);
    acc1 = wmma_f16(a, b1, acc1);
    acc2 = wmma_f16(a, b2, acc2);
  }
  const int hi = lane >> 4;
  const int nc = lane & 15;
#pragma unroll
  for (int v = 0; v < 8; ++v) {
    const int m = m0 + v + 8 * hi;
    float* row = xp2 + (size_t)m * G2;
    __builtin_nontemporal_store(acc0[v] + bih[nc],      &row[nc]);
    __builtin_nontemporal_store(acc1[v] + bih[16 + nc], &row[16 + nc]);
    __builtin_nontemporal_store(acc2[v] + bih[32 + nc], &row[32 + nc]);
  }
}

// ---------------------------------------------------------------------------
// GRU2 recurrence + ReLU + FC + 2*tanh -> x_mid  (single sequential block)
// ---------------------------------------------------------------------------
__global__ void __launch_bounds__(256) k_gru2_fc(
    const float* __restrict xp2,   // (L*B, 48)
    const float* __restrict h2_0,  // (B, 16)
    const float* __restrict whh,   // (48, 16)
    const float* __restrict bhh,   // (48)
    const float* __restrict wfc,   // (20, 16)
    const float* __restrict bfc,   // (20)
    float* __restrict xmid,        // (B, L, 20) -> d_out
    float* __restrict h2f)         // (B, 16)
{
  __shared__ float h[BB * H2];
  __shared__ float y2[BB * H2];
  __shared__ float swhh[G2 * H2];
  __shared__ float swfc[FF * H2];
  __shared__ float sbhh[G2];
  __shared__ float sbfc[FF];
  const int tid = threadIdx.x;
  for (int i = tid; i < BB * H2; i += 256) h[i] = h2_0[i];
  for (int i = tid; i < G2 * H2; i += 256) swhh[i] = whh[i];
  for (int i = tid; i < FF * H2; i += 256) swfc[i] = wfc[i];
  if (tid < G2) sbhh[tid] = bhh[tid];
  if (tid < FF) sbfc[tid] = bfc[tid];
  __syncthreads();

  for (int t = 0; t < LL; ++t) {
    const float* xp = xp2 + (size_t)t * BB * G2;
    if (tid < BB) {
      const int b = tid;
      float* hb = h + b * H2;
      float gh[G2];
      for (int g = 0; g < G2; ++g) {
        const float* w = swhh + g * H2;
        float s = sbhh[g];
#pragma unroll
        for (int k = 0; k < H2; ++k) s += hb[k] * w[k];
        gh[g] = s;
      }
      const float* xb = xp + b * G2;
      float hn[H2];
#pragma unroll
      for (int j = 0; j < H2; ++j) {
        float r = sigmoidf_(xb[j] + gh[j]);
        float z = sigmoidf_(xb[H2 + j] + gh[H2 + j]);
        float n = tanhf(xb[2 * H2 + j] + r * gh[2 * H2 + j]);
        hn[j] = (1.0f - z) * n + z * hb[j];
      }
#pragma unroll
      for (int j = 0; j < H2; ++j) {
        hb[j] = hn[j];
        y2[b * H2 + j] = fmaxf(hn[j], 0.0f);
      }
    }
    __syncthreads();
    for (int idx = tid; idx < BB * FF; idx += 256) {
      const int b = idx / FF, f = idx % FF;
      const float* w = swfc + f * H2;
      const float* yb = y2 + b * H2;
      float s = sbfc[f];
#pragma unroll
      for (int k = 0; k < H2; ++k) s += yb[k] * w[k];
      xmid[((size_t)b * LL + t) * FF + f] = 2.0f * tanhf(s);
    }
    __syncthreads();
  }
  for (int i = tid; i < BB * H2; i += 256) h2f[i] = h[i];
}

// ---------------------------------------------------------------------------
// GRU3 input projection over flipped x_mid: xp3 (L, B, 60)
// ---------------------------------------------------------------------------
__global__ void k_xp3(const float* __restrict xmid, const float* __restrict wih,
                      const float* __restrict bih, float* __restrict xp3) {
  size_t i = blockIdx.x * (size_t)blockDim.x + threadIdx.x;
  const size_t total = (size_t)LL * BB * G3;
  if (i >= total) return;
  const int g = (int)(i % G3);
  const int b = (int)((i / G3) % BB);
  const int t = (int)(i / ((size_t)G3 * BB));
  const float* xv = xmid + ((size_t)b * LL + (LL - 1 - t)) * FF;
  const float* w = wih + g * FF;
  float s = bih[g];
#pragma unroll
  for (int k = 0; k < FF; ++k) s += xv[k] * w[k];
  xp3[i] = s;
}

// ---------------------------------------------------------------------------
// GRU3 recurrence -> x_out = tanh(y3), h3f  (single sequential block)
// ---------------------------------------------------------------------------
__global__ void __launch_bounds__(256) k_gru3(
    const float* __restrict xp3,   // (L, B, 60)
    const float* __restrict h3_0,  // (B, 20)
    const float* __restrict whh,   // (60, 20)
    const float* __restrict bhh,   // (60)
    float* __restrict xout,        // (B, L, 20) -> d_out
    float* __restrict h3f)         // (B, 20)
{
  __shared__ float h[BB * FF];
  __shared__ float swhh[G3 * FF];
  __shared__ float sbhh[G3];
  const int tid = threadIdx.x;
  for (int i = tid; i < BB * FF; i += 256) h[i] = h3_0[i];
  for (int i = tid; i < G3 * FF; i += 256) swhh[i] = whh[i];
  if (tid < G3) sbhh[tid] = bhh[tid];
  __syncthreads();

  for (int t = 0; t < LL; ++t) {
    if (tid < BB) {
      const int b = tid;
      const float* xb = xp3 + ((size_t)t * BB + b) * G3;
      float* hb = h + b * FF;
      float gh[G3];
      for (int g = 0; g < G3; ++g) {
        const float* w = swhh + g * FF;
        float s = sbhh[g];
#pragma unroll
        for (int k = 0; k < FF; ++k) s += hb[k] * w[k];
        gh[g] = s;
      }
      float hn[FF];
#pragma unroll
      for (int j = 0; j < FF; ++j) {
        float r = sigmoidf_(xb[j] + gh[j]);
        float z = sigmoidf_(xb[FF + j] + gh[FF + j]);
        float n = tanhf(xb[2 * FF + j] + r * gh[2 * FF + j]);
        hn[j] = (1.0f - z) * n + z * hb[j];
      }
      float* ob = xout + ((size_t)b * LL + t) * FF;
#pragma unroll
      for (int j = 0; j < FF; ++j) {
        hb[j] = hn[j];
        ob[j] = tanhf(hn[j]);
      }
    }
  }
  __syncthreads();
  for (int i = tid; i < BB * FF; i += 256) h3f[i] = h[i];
}

// ---------------------------------------------------------------------------
// Host-side launcher
// ---------------------------------------------------------------------------
extern "C" void kernel_launch(void* const* d_in, const int* in_sizes, int n_in,
                              void* d_out, int out_size, void* d_ws, size_t ws_size,
                              hipStream_t stream) {
  const float* x    = (const float*)d_in[0];
  const float* h1   = (const float*)d_in[1];
  const float* h2   = (const float*)d_in[2];
  const float* h3   = (const float*)d_in[3];
  const float* Wih1 = (const float*)d_in[4];
  const float* Whh1 = (const float*)d_in[5];
  const float* bih1 = (const float*)d_in[6];
  const float* bhh1 = (const float*)d_in[7];
  const float* Wih2 = (const float*)d_in[8];
  const float* Whh2 = (const float*)d_in[9];
  const float* bih2 = (const float*)d_in[10];
  const float* bhh2 = (const float*)d_in[11];
  const float* Wih3 = (const float*)d_in[12];
  const float* Whh3 = (const float*)d_in[13];
  const float* bih3 = (const float*)d_in[14];
  const float* bhh3 = (const float*)d_in[15];
  const float* Wfc  = (const float*)d_in[16];
  const float* bfc  = (const float*)d_in[17];

  float* out  = (float*)d_out;
  float* xmid = out;                                  // (B,L,20)
  float* xout = xmid + (size_t)BB * LL * FF;          // (B,L,20)
  float* h1f  = xout + (size_t)BB * LL * FF;          // (B,384)
  float* h2f  = h1f + (size_t)BB * H1;                // (B,16)
  float* h3f  = h2f + (size_t)BB * H2;                // (B,20)

  // workspace carve-up (256B aligned)
  uint8_t* ws = (uint8_t*)d_ws;
  size_t off = 0;
  auto carve = [&](size_t bytes) -> uint8_t* {
    uint8_t* p = ws + off;
    off = (off + bytes + 255) & ~(size_t)255;
    return p;
  };
  _Float16* x16    = (_Float16*)carve((size_t)LL * BB * 32 * 2);
  _Float16* w1ih16 = (_Float16*)carve((size_t)G1 * 32 * 2);
  _Float16* w1hh16 = (_Float16*)carve((size_t)G1 * H1 * 2);
  _Float16* w2ih16 = (_Float16*)carve((size_t)G2 * H1 * 2);
  _Float16* h16b   = (_Float16*)carve((size_t)2 * BB * H1 * 2);
  float*    h32b   = (float*)carve((size_t)2 * BB * H1 * 4);
  _Float16* y1     = (_Float16*)carve((size_t)LL * BB * H1 * 2);
  float*    xp2    = (float*)carve((size_t)LL * BB * G2 * 4);
  float*    xp3    = (float*)carve((size_t)LL * BB * G3 * 4);
  unsigned* bar    = (unsigned*)carve(256);

  // 0) init state + barrier
  k_init<<<(BB * H1 + 255) / 256, 256, 0, stream>>>(h1, h32b, h16b, bar);

  // 1) precision/layout conversions
  {
    size_t n = (size_t)LL * BB * 32;
    k_convert_x<<<(unsigned)((n + 255) / 256), 256, 0, stream>>>(x, x16);
  }
  k_convert_w<<<((G1 * 32) + 255) / 256, 256, 0, stream>>>(Wih1, w1ih16, G1, CC, 32);
  k_convert_w<<<((G1 * H1) + 255) / 256, 256, 0, stream>>>(Whh1, w1hh16, G1, H1, H1);
  k_convert_w<<<((G2 * H1) + 255) / 256, 256, 0, stream>>>(Wih2, w2ih16, G2, H1, H1);

  // 2) GRU1 persistent WMMA recurrence (LDS-resident weights)
  k_gru1<<<NB1, 128, 0, stream>>>(x16, w1ih16, w1hh16, bih1, bhh1,
                                  h32b, h16b, y1, h1f, bar);

  // 3) bulk WMMA GEMM: xp2 = y1 * Wih2^T + bih2 (LDS-resident B)
  k_xp2<<<(LL * BB) / 128, 256, 0, stream>>>(y1, w2ih16, bih2, xp2);

  // 4) GRU2 + ReLU + FC -> x_mid
  k_gru2_fc<<<1, 256, 0, stream>>>(xp2, h2, Whh2, bhh2, Wfc, bfc, xmid, h2f);

  // 5) GRU3 input projection on flipped x_mid
  {
    size_t n = (size_t)LL * BB * G3;
    k_xp3<<<(unsigned)((n + 255) / 256), 256, 0, stream>>>(xmid, Wih3, bih3, xp3);
  }

  // 6) GRU3 recurrence -> x_out, h3f
  k_gru3<<<1, 256, 0, stream>>>(xp3, h3, Whh3, bhh3, xout, h3f);
}
